// JumpEulerForwardCuda_79834852098112
// MI455X (gfx1250) — matrix operand order repaired and checked by reading, not compiled, not run on hardware
//
#include <hip/hip_runtime.h>
#include <stdint.h>

// ---------------------------------------------------------------------------
// Jump-diffusion Euler-Maruyama path simulator for MI455X (gfx1250, wave32).
// One wave owns 16 particles for all 200 steps.
//
// Drift MLP (2->64 tanh 64->2) per step:
//   GEMM1 computed TRANSPOSED:  hT = [W1^T | b1] @ [x^T ; 1]
//     M = hidden (4 tiles of 16), N = 16 particles, K = 3 (padded to 32).
//     -> D layout: element j, lane l holds hT[16t + 8*(l/16) + j][l%16],
//        i.e. each lane holds the hidden values of ITS OWN particle row,
//        which is exactly the GEMM2 A-fragment it needs (no LDS transpose).
//   GEMM2: drift = h @ W2 + b2, K = 64 as two chained 16x16x32 f16 WMMAs.
// tanh uses the gfx1250-native v_tanh_f32 trans op when available.
// ---------------------------------------------------------------------------

typedef __attribute__((ext_vector_type(16))) _Float16 v16h;
typedef __attribute__((ext_vector_type(8)))  float    v8f;

#define IN_F   2
#define DIM_H  64
#define NSIM   100000
#define STEPS  200
#define DT        0.02f
#define SQRT_DT   0.1414213562373095f
#define POIS_L    0.449328964117f      /* exp(-INTENSITY*DT) = exp(-0.8) */

#define WAVES_PER_BLOCK 8
#define PARTS_PER_WAVE  16
#define PARTS_PER_BLOCK (WAVES_PER_BLOCK * PARTS_PER_WAVE)   /* 128 */

// ------------------------------ fast tanh -----------------------------------
__device__ __forceinline__ float fast_tanh(float z) {
#if __has_builtin(__builtin_amdgcn_tanhf)
    return __builtin_amdgcn_tanhf(z);       // v_tanh_f32 (gfx1250 trans op)
#else
    float e = __expf(2.0f * z);             // branchless, saturates correctly
    return 1.0f - 2.0f / (e + 1.0f);
#endif
}

// ------------------------- counter-based RNG --------------------------------
__device__ __forceinline__ uint32_t mix32(uint32_t x) {
    x ^= x >> 16; x *= 0x7feb352dU;
    x ^= x >> 15; x *= 0x846ca68bU;
    x ^= x >> 16;
    return x;
}

struct Rng {
    uint32_t s;
    __device__ __forceinline__ Rng(uint32_t p, uint32_t t, uint32_t stream) {
        s = mix32(p * 0x9E3779B9u ^ mix32(t * 0x85EBCA6Bu ^ (stream * 0xC2B2AE35u + 0x27220A95u)));
    }
    __device__ __forceinline__ float u01() {
        s = s * 1664525u + 1013904223u;
        uint32_t r = mix32(s);
        return ((float)(r >> 8) + 0.5f) * (1.0f / 16777216.0f);   // (0,1)
    }
};

__device__ __forceinline__ float normal01(Rng& r) {
    float u1 = r.u01();
    float u2 = r.u01();
    float m  = __builtin_sqrtf(-2.0f * __logf(u1));
    return m * __cosf(6.28318530718f * u2);
}

__device__ __forceinline__ int poisson08(Rng& r) {
    // Knuth: number of uniform draws until product <= exp(-lambda), minus one.
    float p = 1.0f;
    int   k = 0;
    for (int i = 0; i < 16; ++i) {
        p *= r.u01();
        if (p <= POIS_L) break;
        ++k;
    }
    return k;
}

__device__ __forceinline__ float gammaMT(Rng& r, float a) {
    // Marsaglia-Tsang, a >= 1 (concentration is an integer >= 1 here).
    float d = a - (1.0f / 3.0f);
    float c = 0.333333333f * rsqrtf(d);
    for (int i = 0; i < 8; ++i) {
        float x = normal01(r);
        float t = 1.0f + c * x;
        if (t <= 0.0f) continue;
        float v = t * t * t;
        float u = r.u01();
        if (__logf(u) < 0.5f * x * x + d - d * v + d * __logf(v))
            return d * v;
    }
    return d;   // bounded-iteration fallback ~ mode of the distribution
}

// ------------------------------- kernel -------------------------------------
__global__ __launch_bounds__(256)
void jump_euler_wmma_kernel(const float* __restrict__ z0,
                            const float* __restrict__ W1,
                            const float* __restrict__ b1,
                            const float* __restrict__ W2,
                            const float* __restrict__ b2,
                            const float* __restrict__ pdiff,
                            float* __restrict__ out)
{
    // Only LDS use: 32-value drift scatter (D layout -> elementwise lanes).
    __shared__ __align__(16) float dbuf[WAVES_PER_BLOCK][16][IN_F];   // 1 KB

    const int lane = threadIdx.x & 31;
    const int wave = threadIdx.x >> 5;
    const int waveGlobal = blockIdx.x * WAVES_PER_BLOCK + wave;
    const int pbase = waveGlobal * PARTS_PER_WAVE;
    if (pbase >= NSIM) return;                    // whole-wave uniform exit

    const int nl = lane & 15;                     // tile column (N) / particle
    const int hi = lane >> 4;                     // lane half / feature index
    const float diffusion = pdiff[0];

    __builtin_prefetch(W1, 0, 0);                 // global_prefetch_b8
    __builtin_prefetch(W2, 0, 0);

    // ---- constant f16 operands (built once, live in VGPRs) ----
    // GEMM1 A: [W1^T | b1] (64 x 3) zero-padded to (64 x 32) -> 4 M-tiles of
    // 16x32. 16-bit A layout: lane l owns row m = l%16 (+16t); lanes 0-15
    // carry K=0,1 in element 0,1 and K=2 in element 2; lanes 16-31 carry
    // K>=8 which is all padding here.
    v16h aW1[4];
    for (int t = 0; t < 4; ++t) {
        v16h a = {};
        if (hi == 0) {
            int m = 16 * t + nl;                  // hidden unit
            a[0] = (_Float16)W1[0 * DIM_H + m];
            a[1] = (_Float16)W1[1 * DIM_H + m];
            a[2] = (_Float16)b1[m];               // bias folded into K=2 row
        }
        aW1[t] = a;
    }
    // GEMM2 B: W2 (64x2) zero-padded to 64x16 -> two 32x16 K-chunks.
    // 16-bit B layout: lanes 0-15 carry K=0..7 (e 0-7) and K=16..23 (e 8-15),
    // lanes 16-31 carry K=8..15 and K=24..31.
    v16h bW2[2];
    for (int c = 0; c < 2; ++c) {
        v16h b = {};
        if (nl < IN_F) {
            for (int e = 0; e < 16; ++e) {
                int k = c * 32 + 8 * hi + (e < 8 ? e : e + 8);
                b[e] = (_Float16)W2[k * IN_F + nl];
            }
        }
        bW2[c] = b;
    }
    const float bias2 = (nl < IN_F) ? b2[nl] : 0.0f;

    // ---- elementwise state: lane = particle nl, feature hi ----
    const uint32_t P = (uint32_t)(pbase + nl);
    float x = z0[(size_t)P * IN_F + hi];
    out[(size_t)P * (STEPS + 1) * IN_F + hi] = x;                 // t = 0

    for (int s = 0; s < STEPS; ++s) {
        // ------------------- stochastic terms (pure VALU) -------------------
        Rng rp(P, (uint32_t)s, 0u);               // Poisson: feature-shared
        int k = poisson08(rp);
        float jump = 0.0f;
        if (k > 0) {
            Rng rg(P, (uint32_t)s, 1u + (uint32_t)hi);
            float g = gammaMT(rg, (float)k);
            jump = g * (hi == 0 ? 0.1f : 1.0f);   // divide by RATE = [10, 1]
        }
        Rng rn(P, (uint32_t)s, 3u + (uint32_t)hi);
        float noise = normal01(rn);

        // --------------- GEMM1 (transposed): hT = [W1^T|b1] @ [x^T;1] -------
        // B (32x16 f16): lanes 0-15 element {0,1,2} = {x[p][0], x[p][1], 1};
        // lanes 16-31 carry K>=8 -> zero padding.
        float xOther = __shfl_xor(x, 16, 32);
        v16h bx = {};
        if (hi == 0) {
            bx[0] = (_Float16)x;
            bx[1] = (_Float16)xOther;
            bx[2] = (_Float16)1.0f;               // bias row
        }

        // D of tile t: element j, lane l -> hT[16t + 8*hi + j][particle nl].
        // GEMM2 A-fragment (chunk q) for this lane needs exactly
        // K = 16*(2q+u) + 8*hi + j at element 8u+j  ->  tiles feed elements
        // 0..7 (tile 2q) and 8..15 (tile 2q+1) directly. No cross-lane moves.
        v16h A0 = {}, A1 = {};
        for (int t = 0; t < 4; ++t) {
            v8f c = {};
            c = __builtin_amdgcn_wmma_f32_16x16x32_f16(
                    false, aW1[t], false, bx, (short)0, c, false, false);
            for (int j = 0; j < 8; ++j) {
                _Float16 hv = (_Float16)fast_tanh(c[j]);   // v_tanh_f32
                if (t == 0)      A0[j]     = hv;
                else if (t == 1) A0[j + 8] = hv;
                else if (t == 2) A1[j]     = hv;
                else             A1[j + 8] = hv;
            }
        }

        // ------------------- GEMM2: drift = h @ W2 + b2 ---------------------
        v8f c2;
        for (int j = 0; j < 8; ++j) c2[j] = bias2;
        c2 = __builtin_amdgcn_wmma_f32_16x16x32_f16(
                 false, A0, false, bW2[0], (short)0, c2, false, false);
        c2 = __builtin_amdgcn_wmma_f32_16x16x32_f16(
                 false, A1, false, bW2[1], (short)0, c2, false, false);

        // Scatter drift (cols 0,1 only) back to elementwise lanes via LDS.
        // D layout: element j, lane l -> drift[m = j + 8*hi][n = nl].
        if (nl < IN_F) {
            for (int j = 0; j < 8; ++j)
                dbuf[wave][j + 8 * hi][nl] = c2[j];
        }
        __builtin_amdgcn_wave_barrier();   // DS ops within a wave are in-order
        float drift = dbuf[wave][nl][hi];
        __builtin_amdgcn_wave_barrier();   // protect dbuf reuse next iteration

        // ------------------- Euler-Maruyama update + path store -------------
        x += drift * DT + diffusion * SQRT_DT * noise + jump;
        out[(size_t)P * (STEPS + 1) * IN_F + (size_t)(s + 1) * IN_F + hi] = x;
    }
}

// ------------------------------ launcher ------------------------------------
extern "C" void kernel_launch(void* const* d_in, const int* in_sizes, int n_in,
                              void* d_out, int out_size, void* d_ws, size_t ws_size,
                              hipStream_t stream) {
    const float* z0    = (const float*)d_in[0];   // [NSIM, 2]
    const float* W1    = (const float*)d_in[1];   // [2, 64]
    const float* b1    = (const float*)d_in[2];   // [64]
    const float* W2    = (const float*)d_in[3];   // [64, 2]
    const float* b2    = (const float*)d_in[4];   // [2]
    const float* diff  = (const float*)d_in[5];   // scalar
    float*       out   = (float*)d_out;           // [NSIM, 201, 2]

    (void)in_sizes; (void)n_in; (void)out_size; (void)d_ws; (void)ws_size;

    const int blocks = (NSIM + PARTS_PER_BLOCK - 1) / PARTS_PER_BLOCK;  // 782
    jump_euler_wmma_kernel<<<blocks, 256, 0, stream>>>(z0, W1, b1, W2, b2, diff, out);
}
